// VQVAE_9062380995256
// MI455X (gfx1250) — matrix operand order repaired and checked by reading, not compiled, not run on hardware
//
#include <hip/hip_runtime.h>
#include <hip/hip_bf16.h>

// VQ-VAE nearest-codebook search on gfx1250 (MI455X).
// N=65536 z_e rows (D=256), K=4096 codes.
// argmin_k ||z-c_k||^2 == argmin_k (||c_k||^2 - 2 z.c_k)   (||z||^2 is constant per row)
// Cross term: v_wmma_f32_16x16x32_f16 (fp16 in, fp32 accumulate).
//   A = codebook tiles, pre-swizzled fp16 in workspace in exact A-fragment layout,
//       staged into LDS by the Tensor Data Mover (2-slot x 16 KB ring, TENSORcnt).
//   B = 16 z_e rows per wave, held in registers for the whole k loop.
// z_q gathered from the original fp32 codebook (exact).

#define NN 65536
#define DD 256
#define KK 4096

typedef __attribute__((ext_vector_type(16))) _Float16     v16h;
typedef __attribute__((ext_vector_type(8)))  float        v8f;
typedef __attribute__((ext_vector_type(4)))  unsigned int v4u;
typedef __attribute__((ext_vector_type(8)))  int          v8i;
typedef __attribute__((ext_vector_type(4)))  int          v4i;

// ---------------------------------------------------------------------------
// TDM issue: copy one 16 KB staging unit (4096 dwords, 1 row) global -> LDS.
// D# group0: [1:0]=count=1, [63:32]=lds_addr, [120:64]=global_addr, [127:126]=type=2.
// D# group1: data_size=2 (4B), tensor_dim0=tile_dim0=4096, tensor_dim1=tile_dim1=1,
//            tensor_dim0_stride=tensor_dim1_stride=4096. Groups 2/3 zero (<=2D).
// ---------------------------------------------------------------------------
__device__ __forceinline__ void tdm_unit_to_lds(unsigned long long gaddr,
                                                unsigned int ldsOff, v8i g1) {
  v4u g0 = { 1u,                                   // count=1, user descriptor
             ldsOff,                               // lds_addr (bytes)
             (unsigned int)gaddr,                  // global_addr[31:0]
             (unsigned int)(gaddr >> 32) | 0x80000000u };  // [56:32] | type=2<<30
  v4i z4 = {0, 0, 0, 0};
#if __clang_major__ >= 23
  v8i z8 = {0, 0, 0, 0, 0, 0, 0, 0};
  __builtin_amdgcn_tensor_load_to_lds(g0, g1, z4, z4, z8, 0);
#else
  __builtin_amdgcn_tensor_load_to_lds(g0, g1, z4, z4, 0);
#endif
}

// ---------------------------------------------------------------------------
// Kernel 1: codebook f32 -> f16 in WMMA A-fragment layout.
// Fragment gw = kt*8 + dc covers codes [kt*16,+16) x dims [dc*32,+32).
// 16-bit A 16x32 (ISA 7.12.2): lane L<16 -> M=L, halves K{0..7},K{16..23};
// lane L>=16 -> M=L-16, halves K{8..15},K{24..31}. 32 B/lane, 1 KB/fragment.
// ---------------------------------------------------------------------------
__global__ __launch_bounds__(256) void vq_prep_swizzle(const float* __restrict__ cb,
                                                       _Float16* __restrict__ cbw) {
  const int wave = threadIdx.x >> 5;
  const int lane = threadIdx.x & 31;
  const int gw   = blockIdx.x * 8 + wave;      // fragment id 0..2047
  const int kt   = gw >> 3;
  const int dc   = gw & 7;
  const int m    = lane & 15;
  const int kk0  = (lane < 16) ? 0 : 8;

  const float* src = cb + (size_t)(kt * 16 + m) * DD + dc * 32 + kk0;
  v16h out;
#pragma unroll
  for (int j = 0; j < 8; ++j) out[j]     = (_Float16)src[j];
#pragma unroll
  for (int j = 0; j < 8; ++j) out[8 + j] = (_Float16)src[16 + j];
  *(v16h*)(cbw + (size_t)gw * 512 + lane * 16) = out;
}

// ---------------------------------------------------------------------------
// Kernel 2: cb_sq[k] = sum_d cb[k][d]^2 (fp32). One wave per code; ds_swizzle
// xor-tree reduction (group-of-32 encoding: imm = (xor<<10) | 0x1F).
// ---------------------------------------------------------------------------
__global__ __launch_bounds__(256) void vq_prep_sq(const float* __restrict__ cb,
                                                  float* __restrict__ cbsq) {
  const int wave = threadIdx.x >> 5;
  const int lane = threadIdx.x & 31;
  const int k    = blockIdx.x * 8 + wave;

  const float4* p = (const float4*)(cb + (size_t)k * DD + lane * 8);
  float4 a = p[0], b = p[1];
  float s = a.x * a.x + a.y * a.y + a.z * a.z + a.w * a.w
          + b.x * b.x + b.y * b.y + b.z * b.z + b.w * b.w;
  s += __builtin_bit_cast(float, __builtin_amdgcn_ds_swizzle(__builtin_bit_cast(int, s), 0x041F));
  s += __builtin_bit_cast(float, __builtin_amdgcn_ds_swizzle(__builtin_bit_cast(int, s), 0x081F));
  s += __builtin_bit_cast(float, __builtin_amdgcn_ds_swizzle(__builtin_bit_cast(int, s), 0x101F));
  s += __builtin_bit_cast(float, __builtin_amdgcn_ds_swizzle(__builtin_bit_cast(int, s), 0x201F));
  s += __builtin_bit_cast(float, __builtin_amdgcn_ds_swizzle(__builtin_bit_cast(int, s), 0x401F));
  if (lane == 0) cbsq[k] = s;
}

// ---------------------------------------------------------------------------
// Kernel 3: main. Block = 8 waves, 128 z_e rows. Wave 0 drives the TDM:
// 16 KB staging units (2 k-tiles each), 2-slot LDS ring, prefetch unit u+1
// while all waves consume unit u. Per k-tile: stage 8 A fragments from LDS
// into registers (one ds clause), 8 chained WMMAs, fuse ||c||^2 - 2*cross,
// per-lane running argmin; merge half-waves with ds_swizzle SWAPX16; gather z_q.
// ---------------------------------------------------------------------------
__global__ __launch_bounds__(256) void vq_main(const float* __restrict__ ze,
                                               const float* __restrict__ cb,
                                               const _Float16* __restrict__ cbw,
                                               const float* __restrict__ cbsq,
                                               float* __restrict__ zq,
                                               int* __restrict__ idxOut) {
  __shared__ __align__(1024) char smemA[2 * 16384];   // TDM ring: 2 x 16 KB units
  __shared__ int ldsIdx[128];

  const int wave  = threadIdx.x >> 5;
  const int lane  = threadIdx.x & 31;
  const int nBase = blockIdx.x * 128;
  const int col   = lane & 15;
  const int doff  = (lane < 16) ? 0 : 8;
  const int n     = nBase + wave * 16 + col;   // this lane's z_e row (col of B)

  // TDM group1 descriptor (loop-invariant): data_size=4B, dim0=4096, dim1=1,
  // tile0=4096, tile1=1, stride0=stride1=4096.
  const v8i g1 = { 0x00020000, 0x10000000, 0x00010000, 0x10000000,
                   0x00000001, 4096,       0x10000000, 0 };
  const unsigned int       ldsBase = (unsigned int)(unsigned long long)(uintptr_t)smemA;
  const unsigned long long cbwAddr = (unsigned long long)(uintptr_t)cbw;

  // Preload unit 0 (tiles 0,1); overlap with B-fragment construction below.
  if (wave == 0) tdm_unit_to_lds(cbwAddr, ldsBase, g1);

  // Load + convert this wave's 16 z_e rows into B fragments (registers).
  // B 32x16 f16: lane L<16 -> col N=L, d {0..7,16..23}; L>=16 -> d {8..15,24..31}.
  v16h B[8];
  const float* zp = ze + (size_t)n * DD + doff;
#pragma unroll
  for (int f = 0; f < 8; ++f) {
    const float4* q = (const float4*)(zp + f * 32);
    float4 x0 = q[0], x1 = q[1], x2 = q[4], x3 = q[5];
    v16h b;
    b[0]  = (_Float16)x0.x; b[1]  = (_Float16)x0.y; b[2]  = (_Float16)x0.z; b[3]  = (_Float16)x0.w;
    b[4]  = (_Float16)x1.x; b[5]  = (_Float16)x1.y; b[6]  = (_Float16)x1.z; b[7]  = (_Float16)x1.w;
    b[8]  = (_Float16)x2.x; b[9]  = (_Float16)x2.y; b[10] = (_Float16)x2.z; b[11] = (_Float16)x2.w;
    b[12] = (_Float16)x3.x; b[13] = (_Float16)x3.y; b[14] = (_Float16)x3.z; b[15] = (_Float16)x3.w;
    B[f] = b;
  }

  if (wave == 0) __builtin_amdgcn_s_wait_tensorcnt(0);  // unit 0 resident
  __syncthreads();

  const int kk0    = (lane < 16) ? 0 : 8;  // C layout: lane<16 -> M=j, else M=8+j
  const int laneOff = lane * 32;
  float minV = 3.4e38f;
  int   minI = 0;

  for (int u = 0; u < 128; ++u) {          // 128 units x 2 k-tiles = 256 k-tiles
    // Prefetch next unit into the other ring slot (its last readers finished
    // at iteration u-1; the barrier there fences the overwrite).
    if (wave == 0 && u < 127)
      tdm_unit_to_lds(cbwAddr + (unsigned long long)(u + 1) * 16384,
                      ldsBase + (unsigned int)(((u + 1) & 1) << 14), g1);

    const char* base = smemA + ((u & 1) << 14);
#pragma unroll
    for (int t = 0; t < 2; ++t) {
      const char* ap = base + t * 8192;
      // Stage all 8 A fragments first: one ds_load_b128 clause, graduated waits.
      v16h a[8];
#pragma unroll
      for (int dc = 0; dc < 8; ++dc)
        a[dc] = *(const v16h*)(ap + dc * 1024 + laneOff);

      v8f c = {0.f, 0.f, 0.f, 0.f, 0.f, 0.f, 0.f, 0.f};
#pragma unroll
      for (int dc = 0; dc < 8; ++dc)
        c = __builtin_amdgcn_wmma_f32_16x16x32_f16(false, a[dc], false, B[dc],
                                                   (short)0, c, false, false);

      const int kbase = u * 32 + t * 16 + kk0;
      const float4* sp = (const float4*)(cbsq + kbase);
      float4 s0 = sp[0], s1 = sp[1];
      float sv[8] = {s0.x, s0.y, s0.z, s0.w, s1.x, s1.y, s1.z, s1.w};
#pragma unroll
      for (int j = 0; j < 8; ++j) {
        float v = sv[j] - 2.0f * c[j];          // + ||z||^2 omitted (row-constant)
        if (v < minV) { minV = v; minI = kbase + j; }  // scan order keeps first idx
      }
    }

    if (wave == 0) __builtin_amdgcn_s_wait_tensorcnt(0);  // next unit resident
    __syncthreads();
  }

  // Merge lanes L and L^16 (same column n, complementary k subsets): SWAPX16.
  float oV = __builtin_bit_cast(float,
               __builtin_amdgcn_ds_swizzle(__builtin_bit_cast(int, minV), 0x401F));
  int   oI = __builtin_amdgcn_ds_swizzle(minI, 0x401F);
  if (oV < minV || (oV == minV && oI < minI)) { minV = oV; minI = oI; }

  if (lane < 16) {
    idxOut[n] = minI;
    ldsIdx[wave * 16 + col] = minI;
  }
  __syncthreads();

  // Gather z_q = codebook[idx] in fp32 (exact), block-cooperative, coalesced.
  for (int i = threadIdx.x; i < 128 * 64; i += 256) {
    const int row = i >> 6;
    const int c4  = i & 63;
    const int idx = ldsIdx[row];
    float4 v = ((const float4*)(cb + (size_t)idx * DD))[c4];
    ((float4*)(zq + (size_t)(nBase + row) * DD))[c4] = v;
  }
}

// ---------------------------------------------------------------------------
extern "C" void kernel_launch(void* const* d_in, const int* in_sizes, int n_in,
                              void* d_out, int out_size, void* d_ws, size_t ws_size,
                              hipStream_t stream) {
  const float* z_e = (const float*)d_in[0];   // [N, D] f32
  const float* cb  = (const float*)d_in[1];   // [K, D] f32

  _Float16* cbw  = (_Float16*)d_ws;                                            // 2 MB swizzled f16
  float*    cbsq = (float*)((char*)d_ws + (size_t)KK * DD * sizeof(_Float16)); // 16 KB

  float* zq     = (float*)d_out;                    // [N, D] f32
  int*   idxOut = (int*)(zq + (size_t)NN * DD);     // [N] int32 (tuple tail)

  vq_prep_swizzle<<<256, 256, 0, stream>>>(cb, cbw);
  vq_prep_sq<<<512, 256, 0, stream>>>(cb, cbsq);
  vq_main<<<NN / 128, 256, 0, stream>>>(z_e, cb, cbw, cbsq, zq, idxOut);
}
